// Interest_11759620456477
// MI455X (gfx1250) — compile-verified, matrix operand order checked by reading, not run on hardware
//
#include <hip/hip_runtime.h>
#include <hip/hip_bf16.h>

// ---------------------------------------------------------------------------
// DIEN forward for MI455X (gfx1250, wave32, WMMA, 320KB LDS/WGP).
// B=1024, T=200, D=128. Recurrences fused: weights resident in LDS as bf16
// (loaded via TDM tensor_load_to_lds where available), h stays resident,
// all matmuls use v_wmma_f32_16x16x32_bf16 with f32 accumulation.
// ---------------------------------------------------------------------------

#define Bb 1024
#define Tt 200
#define Dd 128
#define N3 384   // 3*D

typedef __attribute__((ext_vector_type(16))) __bf16 v16bf;
typedef __attribute__((ext_vector_type(8)))  __bf16 v8bf;
typedef __attribute__((ext_vector_type(4)))  __bf16 v4bf;
typedef __attribute__((ext_vector_type(2)))  __bf16 v2bf;
typedef __attribute__((ext_vector_type(8)))  float  v8f;
typedef __attribute__((ext_vector_type(4)))  unsigned int v4u;
typedef __attribute__((ext_vector_type(8)))  int v8i;
typedef __attribute__((ext_vector_type(4)))  int v4i;

__device__ __forceinline__ float sigf(float x) { return 1.f / (1.f + __expf(-x)); }

// Load a 16x32 bf16 fragment from LDS (row-major, leading dim `ld`).
// CDNA5 16-bit A layout: lanes 0-15 = rows, K in {0..7,16..23}; lanes 16-31
// hold K in {8..15,24..31}.  Two ds_load_b128 per fragment.
__device__ __forceinline__ v16bf ldfrag(const __bf16* base, int ld, int k0) {
  int lane = threadIdx.x & 31;
  int row  = lane & 15;
  int hi   = lane >> 4;
  const __bf16* p0 = base + row * ld + k0 + hi * 8;
  v8bf a = *(const v8bf*)(p0);
  v8bf b = *(const v8bf*)(p0 + 16);
  return __builtin_shufflevector(a, b, 0,1,2,3,4,5,6,7,8,9,10,11,12,13,14,15);
}

__device__ __forceinline__ v8f wmma_bf16(v16bf a, v16bf b, v8f c) {
  // emits v_wmma_f32_16x16x32_bf16
  return __builtin_amdgcn_wmma_f32_16x16x32_bf16(false, a, false, b, (short)0, c,
                                                 false, false);
}

// Store a 16x16 f32 C/D fragment to LDS: VGPR i, lanes 0-15 -> (M=i, N=lane);
// lanes 16-31 -> (M=i+8, N=lane-16).
__device__ __forceinline__ void stfrag(float* base, int ld, v8f c) {
  int lane = threadIdx.x & 31;
  int col  = lane & 15;
  int mo   = (lane >> 4) * 8;
#pragma unroll
  for (int i = 0; i < 8; ++i) base[(mo + i) * ld + col] = c[i];
}

// ---------------------------------------------------------------------------
// TDM bulk LDS fill: 1-D bf16 tensor descriptor (D# groups per ISA ch.8),
// issued once by wave 0; falls back to vectorized ds copies when the builtin
// is unavailable (host pass / older toolchains).
// ---------------------------------------------------------------------------
__device__ __forceinline__ void tdm_fill(__bf16* dst, const __bf16* src,
                                         unsigned elems, int tid, int nthr) {
#if __has_builtin(__builtin_amdgcn_tensor_load_to_lds)
  (void)nthr;
  if (tid < 32) {                          // one wave issues the DMA
    unsigned lo = (unsigned)(size_t)(void*)dst;            // LDS byte offset
    unsigned long long ga = (unsigned long long)(size_t)(const void*)src;
    // group0: count=1 | lds_addr | global_addr[56:0] | type=2 ("image")
    v4u g0 = { 1u, lo, (unsigned)ga,
               (unsigned)((ga >> 32) & 0x1FFFFFFu) | (2u << 30) };
    // group1: data_size=2B; tensor_dim0=elems, tensor_dim1=1;
    //         tile_dim0=elems (1-D), tile_dim1=1; stride0=elems.
    v8i g1 = { (int)0x00010000,
               (int)((elems & 0xFFFFu) << 16),
               (int)((elems >> 16) | (1u << 16)),
               (int)((elems & 0xFFFFu) << 16),
               1, (int)elems, 0, 0 };
    v4i gz = { 0, 0, 0, 0 };
#if __clang_major__ >= 23
    v8i gz8 = { 0, 0, 0, 0, 0, 0, 0, 0 };
    __builtin_amdgcn_tensor_load_to_lds(g0, g1, gz, gz, gz8, 0);
#else
    __builtin_amdgcn_tensor_load_to_lds(g0, g1, gz, gz, 0);
#endif
  }
#else
  for (int i = tid * 8; i < (int)elems; i += nthr * 8)
    *(v8bf*)(dst + i) = *(const v8bf*)(src + i);
#endif
}

__device__ __forceinline__ void tdm_wait(int tid) {
#if __has_builtin(__builtin_amdgcn_s_wait_tensorcnt) && \
    __has_builtin(__builtin_amdgcn_tensor_load_to_lds)
  if (tid < 32) __builtin_amdgcn_s_wait_tensorcnt(0);
#else
  (void)tid;
#endif
}

// ---------------------------------------------------------------------------
// Weight packing: f32 -> bf16, optionally transposed to [N][K] layout.
// ---------------------------------------------------------------------------
__global__ void pack_conv(__bf16* dst, const float* src, int n) {
  int i = blockIdx.x * 256 + threadIdx.x;
  if (i < n) dst[i] = (__bf16)src[i];
}

// dst[n*K + k] = src[k*N + n]   (src is (K,N) row-major)
__global__ void pack_tr(__bf16* dst, const float* src, int K, int N) {
  int i = blockIdx.x * 256 + threadIdx.x;
  if (i < K * N) {
    int n = i / K, k = i - n * K;
    dst[i] = (__bf16)src[k * N + n];
  }
}

__global__ void init_accum(float* a) {
  if (threadIdx.x == 0 && blockIdx.x == 0) a[0] = 0.f;
}

// ---------------------------------------------------------------------------
// Fused GRU over all T steps.  One WG = 16 batch rows, 256 threads (8 waves).
// LDS: W_ih + W_hh packs (bf16, TDM-loaded) + x/h tiles + f32 scratch = 256KB.
// ---------------------------------------------------------------------------
__global__ void __launch_bounds__(256)
gru_fused(const float* __restrict__ keys, const int* __restrict__ lens,
          const __bf16* __restrict__ wih_g, const __bf16* __restrict__ whh_g,
          const float* __restrict__ bih, const float* __restrict__ bhh,
          float* __restrict__ interests) {
  extern __shared__ __align__(16) char smem[];
  __bf16* wih = (__bf16*)smem;            // 384*128 bf16
  __bf16* whh = wih + N3 * Dd;            // 384*128 bf16
  __bf16* xbf = whh + N3 * Dd;            // 16*128 bf16
  __bf16* hbf = xbf + 16 * Dd;            // 16*128 bf16
  float*  gi  = (float*)(hbf + 16 * Dd);  // 16*384 f32
  float*  gh  = gi + 16 * N3;             // 16*384 f32
  float*  hs  = gh + 16 * N3;             // 16*128 f32

  const int tid  = threadIdx.x;
  const int wave = tid >> 5;
  const int b0   = blockIdx.x * 16;

  tdm_fill(wih, wih_g, N3 * Dd, tid, 256);
  tdm_fill(whh, whh_g, N3 * Dd, tid, 256);

  int lenp[4];
#pragma unroll
  for (int i = 0; i < 4; ++i) {            // pairwise h init
    int e = (tid + i * 256) * 2;
    lenp[i] = lens[b0 + (e >> 7)];
    *(float2*)&hs[e] = make_float2(0.f, 0.f);
    *(v2bf*)(hbf + e) = v2bf{(__bf16)0.f, (__bf16)0.f};
  }
  tdm_wait(tid);
  __syncthreads();

  for (int t = 0; t < Tt; ++t) {
#pragma unroll
    for (int i = 0; i < 2; ++i) {          // x tile: 512 float4, 2 per thread
      int q = tid + i * 256;
      int r = q >> 5, d4 = (q & 31) * 4;
      const float* gp = &keys[((size_t)(b0 + r) * Tt + t) * Dd + d4];
      float4 xv = *(const float4*)gp;
      *(v4bf*)(xbf + r * Dd + d4) =
          v4bf{(__bf16)xv.x, (__bf16)xv.y, (__bf16)xv.z, (__bf16)xv.w};
      if (t + 1 < Tt) __builtin_prefetch(gp + Dd, 0, 0);   // global_prefetch_b8
    }
    __syncthreads();
#pragma unroll
    for (int j = 0; j < 3; ++j) {          // 8 waves * 3 = 24 N-tiles
      int nt = wave * 3 + j;
      v8f ai = {}, ah = {};
#pragma unroll
      for (int kk = 0; kk < 4; ++kk) {
        ai = wmma_bf16(ldfrag(xbf, Dd, kk * 32),
                       ldfrag(wih + nt * 16 * Dd, Dd, kk * 32), ai);
        ah = wmma_bf16(ldfrag(hbf, Dd, kk * 32),
                       ldfrag(whh + nt * 16 * Dd, Dd, kk * 32), ah);
      }
      stfrag(gi + nt * 16, N3, ai);
      stfrag(gh + nt * 16, N3, ah);
    }
    __syncthreads();
#pragma unroll
    for (int i = 0; i < 4; ++i) {          // gate combine, 1024 pairs
      int e = (tid + i * 256) * 2, r = e >> 7, d = e & 127;
      float2 ir  = *(const float2*)&gi[r * N3 + d];
      float2 iz  = *(const float2*)&gi[r * N3 + 128 + d];
      float2 in_ = *(const float2*)&gi[r * N3 + 256 + d];
      float2 hr  = *(const float2*)&gh[r * N3 + d];
      float2 hz  = *(const float2*)&gh[r * N3 + 128 + d];
      float2 hn  = *(const float2*)&gh[r * N3 + 256 + d];
      float2 br_ = *(const float2*)&bih[d];
      float2 bz_ = *(const float2*)&bih[128 + d];
      float2 bn_ = *(const float2*)&bih[256 + d];
      float2 cr  = *(const float2*)&bhh[d];
      float2 cz  = *(const float2*)&bhh[128 + d];
      float2 cn  = *(const float2*)&bhh[256 + d];
      float2 h0  = *(const float2*)&hs[e];
      float rg0 = sigf(ir.x + br_.x + hr.x + cr.x);
      float rg1 = sigf(ir.y + br_.y + hr.y + cr.y);
      float zg0 = sigf(iz.x + bz_.x + hz.x + cz.x);
      float zg1 = sigf(iz.y + bz_.y + hz.y + cz.y);
      float ng0 = tanhf(in_.x + bn_.x + rg0 * (hn.x + cn.x));
      float ng1 = tanhf(in_.y + bn_.y + rg1 * (hn.y + cn.y));
      float h2a = (1.f - zg0) * ng0 + zg0 * h0.x;
      float h2b = (1.f - zg1) * ng1 + zg1 * h0.y;
      *(float2*)&hs[e] = make_float2(h2a, h2b);
      *(v2bf*)(hbf + e) = v2bf{(__bf16)h2a, (__bf16)h2b};
      bool m = t < lenp[i];
      *(float2*)&interests[((size_t)(b0 + r) * Tt + t) * Dd + d] =
          make_float2(m ? h2a : 0.f, m ? h2b : 0.f);
    }
    __syncthreads();
  }
}

// ---------------------------------------------------------------------------
// DIN attention -> masked softmax scores.  One WG per batch row.
// ---------------------------------------------------------------------------
__global__ void __launch_bounds__(256)
attention_k(const float* __restrict__ query, const float* __restrict__ st,
            const int* __restrict__ lens,
            const float* __restrict__ W0, const float* __restrict__ b0,
            const float* __restrict__ a0, const float* __restrict__ W1,
            const float* __restrict__ b1, const float* __restrict__ a1,
            const float* __restrict__ W2, const float* __restrict__ b2,
            float* __restrict__ scores) {
  __shared__ float qs[Dd];
  __shared__ float w0s[512 * 16];
  __shared__ float red[256];
  int b = blockIdx.x, tid = threadIdx.x;
  if (tid < Dd) qs[tid] = query[b * Dd + tid];
  for (int i = tid; i < 512 * 16; i += 256) w0s[i] = W0[i];
  __syncthreads();

  int len = lens[b];
  float al0 = a0[0], al1 = a1[0];
  float sc = -3.0e38f;
  if (tid < Tt) {
    const float* sp = st + ((size_t)b * Tt + tid) * Dd;
    float h0[16];
#pragma unroll
    for (int j = 0; j < 16; ++j) h0[j] = b0[j];
    for (int k = 0; k < Dd; ++k) {
      float qv = qs[k], sv = sp[k];
      float f2 = qv - sv, f3 = qv * sv;
      const float* w0a = w0s + k * 16;
      const float* w0b = w0s + (128 + k) * 16;
      const float* w0c = w0s + (256 + k) * 16;
      const float* w0d = w0s + (384 + k) * 16;
#pragma unroll
      for (int j = 0; j < 16; ++j)
        h0[j] += qv * w0a[j] + sv * w0b[j] + f2 * w0c[j] + f3 * w0d[j];
    }
#pragma unroll
    for (int j = 0; j < 16; ++j) h0[j] = h0[j] >= 0.f ? h0[j] : al0 * h0[j];
    float h1[8];
#pragma unroll
    for (int i = 0; i < 8; ++i) {
      float a = b1[i];
#pragma unroll
      for (int j = 0; j < 16; ++j) a += h0[j] * W1[j * 8 + i];
      h1[i] = a >= 0.f ? a : al1 * a;
    }
    float s1 = b2[0];
#pragma unroll
    for (int i = 0; i < 8; ++i) s1 += h1[i] * W2[i];
    sc = (tid < len) ? s1 : -1e9f;
  }
  red[tid] = sc; __syncthreads();
  for (int off = 128; off > 0; off >>= 1) {
    if (tid < off) red[tid] = fmaxf(red[tid], red[tid + off]);
    __syncthreads();
  }
  float mx = red[0]; __syncthreads();
  float ev = (tid < Tt) ? __expf(sc - mx) : 0.f;
  red[tid] = ev; __syncthreads();
  for (int off = 128; off > 0; off >>= 1) {
    if (tid < off) red[tid] += red[tid + off];
    __syncthreads();
  }
  float inv = 1.f / red[0];
  if (tid < Tt) scores[(size_t)b * Tt + tid] = ev * inv;
}

// ---------------------------------------------------------------------------
// Auxiliary loss.  One WG = 16 rows of (B*(T-1)), 128 threads (4 waves).
// Layer0 (256->64) via WMMA, layers 64->32->1 in VALU, masked BCE -> atomic.
// ---------------------------------------------------------------------------
__global__ void __launch_bounds__(128)
aux_k(const float* __restrict__ st, const float* __restrict__ keys,
      const float* __restrict__ negk, const int* __restrict__ lens,
      const __bf16* __restrict__ w0g,
      const float* __restrict__ b0, const float* __restrict__ W1,
      const float* __restrict__ b1, const float* __restrict__ W2,
      const float* __restrict__ b2, float* __restrict__ accum) {
  __shared__ __align__(16) __bf16 w0s[64 * 256];
  __shared__ __align__(16) __bf16 abf[16 * 256];
  __shared__ float h0s[16 * 64];
  __shared__ float lsum[16];
  int tid = threadIdx.x, wave = tid >> 5;
  int m0 = blockIdx.x * 16;
  tdm_fill(w0s, w0g, 64 * 256, tid, 128);
  tdm_wait(tid);

  float myloss = 0.f;
  for (int pass = 0; pass < 2; ++pass) {
    const float* nx = pass ? negk : keys;
    __syncthreads();
#pragma unroll
    for (int i = 0; i < 16; ++i) {          // 2048 bf16 pairs
      int e = (tid + i * 128) * 2, r = e >> 8, k = e & 255;
      int m = m0 + r, b = m / (Tt - 1), t = m - b * (Tt - 1);
      const float* src = (k < Dd)
          ? &st[((size_t)b * Tt + t) * Dd + k]
          : &nx[((size_t)b * Tt + (t + 1)) * Dd + (k - Dd)];
      float2 v = *(const float2*)src;
      *(v2bf*)(abf + e) = v2bf{(__bf16)v.x, (__bf16)v.y};
    }
    __syncthreads();
    {
      v8f acc = {};
#pragma unroll
      for (int kk = 0; kk < 8; ++kk)
        acc = wmma_bf16(ldfrag(abf, 256, kk * 32),
                        ldfrag(w0s + wave * 16 * 256, 256, kk * 32), acc);
      stfrag(h0s + wave * 16, 64, acc);
    }
    __syncthreads();
    if (tid < 16) {
      int r = tid, m = m0 + r, b = m / (Tt - 1), t = m - b * (Tt - 1);
      if (t < lens[b] - 1) {
        float h0v[64];
#pragma unroll
        for (int j = 0; j < 64; ++j) h0v[j] = sigf(h0s[r * 64 + j] + b0[j]);
        float h1[32];
        for (int i2 = 0; i2 < 32; ++i2) {
          float a = b1[i2];
          for (int j = 0; j < 64; ++j) a += h0v[j] * W1[j * 32 + i2];
          h1[i2] = sigf(a);
        }
        float s2 = b2[0];
        for (int i2 = 0; i2 < 32; ++i2) s2 += h1[i2] * W2[i2];
        float p = sigf(s2);
        const float eps = 1e-12f;
        myloss += (pass == 0) ? -logf(fmaxf(p, eps)) : -logf(fmaxf(1.f - p, eps));
      }
    }
  }
  __syncthreads();
  if (tid < 16) lsum[tid] = myloss;
  __syncthreads();
  if (tid == 0) {
    float s = 0.f;
    for (int i = 0; i < 16; ++i) s += lsum[i];
    atomicAdd(accum, s);
  }
}

// ---------------------------------------------------------------------------
// Fused AUGRU over all T steps.  One WG = 16 batch rows, 8 waves.
// ---------------------------------------------------------------------------
__global__ void __launch_bounds__(256)
augru_fused(const float* __restrict__ xin, const float* __restrict__ scores,
            const int* __restrict__ lens,
            const __bf16* __restrict__ wx_g, const __bf16* __restrict__ whrz_g,
            const __bf16* __restrict__ whh_g,
            const float* __restrict__ br, const float* __restrict__ bz,
            const float* __restrict__ bh, float* __restrict__ outp) {
  extern __shared__ __align__(16) char smem[];
  __bf16* wx   = (__bf16*)smem;              // 384*128
  __bf16* whrz = wx + N3 * Dd;               // 256*128
  __bf16* whh  = whrz + 256 * Dd;            // 128*128
  __bf16* xbf  = whh + Dd * Dd;              // 16*128
  __bf16* hbf  = xbf + 16 * Dd;              // 16*128
  __bf16* rhbf = hbf + 16 * Dd;              // 16*128
  float*  gx   = (float*)(rhbf + 16 * Dd);   // 16*384
  float*  gh   = gx + 16 * N3;               // 16*256 (reused 16*128 phase2)
  float*  hs   = gh + 16 * 256;              // 16*128
  float*  zs   = hs + 16 * Dd;               // 16*128

  const int tid = threadIdx.x, wave = tid >> 5, b0 = blockIdx.x * 16;

  tdm_fill(wx,   wx_g,   N3 * Dd,  tid, 256);
  tdm_fill(whrz, whrz_g, 256 * Dd, tid, 256);
  tdm_fill(whh,  whh_g,  Dd * Dd,  tid, 256);

  int lenp[4];
#pragma unroll
  for (int i = 0; i < 4; ++i) {
    int e = (tid + i * 256) * 2;
    lenp[i] = lens[b0 + (e >> 7)];
    *(float2*)&hs[e] = make_float2(0.f, 0.f);
    *(v2bf*)(hbf + e) = v2bf{(__bf16)0.f, (__bf16)0.f};
  }
  tdm_wait(tid);
  __syncthreads();

  for (int t = 0; t < Tt; ++t) {
#pragma unroll
    for (int i = 0; i < 2; ++i) {          // x tile: 512 float4
      int q = tid + i * 256;
      int r = q >> 5, d4 = (q & 31) * 4;
      const float* gp = &xin[((size_t)(b0 + r) * Tt + t) * Dd + d4];
      float4 xv = *(const float4*)gp;
      *(v4bf*)(xbf + r * Dd + d4) =
          v4bf{(__bf16)xv.x, (__bf16)xv.y, (__bf16)xv.z, (__bf16)xv.w};
      if (t + 1 < Tt) __builtin_prefetch(gp + Dd, 0, 0);
    }
    __syncthreads();
#pragma unroll
    for (int j = 0; j < 5; ++j) {          // 8 waves * 5 = 40 tile-jobs
      int job = wave * 5 + j;              // wave-uniform -> EXEC stays full
      v8f acc = {};
      if (job < 24) {                      // gx = x @ [wxr|wxz|wxh]
#pragma unroll
        for (int kk = 0; kk < 4; ++kk)
          acc = wmma_bf16(ldfrag(xbf, Dd, kk * 32),
                          ldfrag(wx + job * 16 * Dd, Dd, kk * 32), acc);
        stfrag(gx + job * 16, N3, acc);
      } else {                             // gh = h @ [whr|whz]
        int nt = job - 24;
#pragma unroll
        for (int kk = 0; kk < 4; ++kk)
          acc = wmma_bf16(ldfrag(hbf, Dd, kk * 32),
                          ldfrag(whrz + nt * 16 * Dd, Dd, kk * 32), acc);
        stfrag(gh + nt * 16, 256, acc);
      }
    }
    __syncthreads();
#pragma unroll
    for (int i = 0; i < 4; ++i) {          // r, z (attention-scaled), pairs
      int e = (tid + i * 256) * 2, r = e >> 7, d = e & 127;
      float at = scores[(size_t)(b0 + (e >> 7)) * Tt + t];
      float2 gxr = *(const float2*)&gx[r * N3 + d];
      float2 gxz = *(const float2*)&gx[r * N3 + 128 + d];
      float2 ghr = *(const float2*)&gh[r * 256 + d];
      float2 ghz = *(const float2*)&gh[r * 256 + 128 + d];
      float2 br_ = *(const float2*)&br[d];
      float2 bz_ = *(const float2*)&bz[d];
      float2 h0  = *(const float2*)&hs[e];
      float rg0 = sigf(gxr.x + ghr.x + br_.x);
      float rg1 = sigf(gxr.y + ghr.y + br_.y);
      float zg0 = at * sigf(gxz.x + ghz.x + bz_.x);
      float zg1 = at * sigf(gxz.y + ghz.y + bz_.y);
      *(float2*)&zs[e] = make_float2(zg0, zg1);
      *(v2bf*)(rhbf + e) = v2bf{(__bf16)(rg0 * h0.x), (__bf16)(rg1 * h0.y)};
    }
    __syncthreads();
    {                                       // h_tilde matmul: (r*h) @ whh
      int nt = wave;                        // 8 tiles, one per wave
      v8f acc = {};
#pragma unroll
      for (int kk = 0; kk < 4; ++kk)
        acc = wmma_bf16(ldfrag(rhbf, Dd, kk * 32),
                        ldfrag(whh + nt * 16 * Dd, Dd, kk * 32), acc);
      stfrag(gh + nt * 16, Dd, acc);        // reuse gh as [16][128]
    }
    __syncthreads();
#pragma unroll
    for (int i = 0; i < 4; ++i) {           // masked carry update, pairs
      int e = (tid + i * 256) * 2, r = e >> 7, d = e & 127;
      float2 gxn = *(const float2*)&gx[r * N3 + 256 + d];
      float2 ghn = *(const float2*)&gh[r * Dd + d];
      float2 bh_ = *(const float2*)&bh[d];
      float2 h0  = *(const float2*)&hs[e];
      float2 zg  = *(const float2*)&zs[e];
      float ht0 = tanhf(gxn.x + ghn.x + bh_.x);
      float ht1 = tanhf(gxn.y + ghn.y + bh_.y);
      float hy0 = (1.f - zg.x) * h0.x + zg.x * ht0;
      float hy1 = (1.f - zg.y) * h0.y + zg.y * ht1;
      bool m = t < lenp[i];
      float h2a = m ? hy0 : h0.x;
      float h2b = m ? hy1 : h0.y;
      *(float2*)&hs[e] = make_float2(h2a, h2b);
      *(v2bf*)(hbf + e) = v2bf{(__bf16)h2a, (__bf16)h2b};
    }
    __syncthreads();
  }
#pragma unroll
  for (int i = 0; i < 4; ++i) {
    int e = (tid + i * 256) * 2, r = e >> 7, d = e & 127;
    *(float2*)&outp[(size_t)(b0 + r) * Dd + d] = *(const float2*)&hs[e];
  }
}

__global__ void finalize_k(const float* __restrict__ accum,
                           const int* __restrict__ lens,
                           float* __restrict__ outloss) {
  if (threadIdx.x == 0 && blockIdx.x == 0) {
    long long n = 0;
    for (int b = 0; b < Bb; ++b) n += (long long)(lens[b] - 1);
    outloss[0] = accum[0] / (2.0f * (float)n);
  }
}

// ---------------------------------------------------------------------------
extern "C" void kernel_launch(void* const* d_in, const int* in_sizes, int n_in,
                              void* d_out, int out_size, void* d_ws, size_t ws_size,
                              hipStream_t stream) {
  (void)in_sizes; (void)n_in; (void)out_size; (void)ws_size;
  // setup_inputs() flattening order:
  const float* query = (const float*)d_in[0];
  const float* keys  = (const float*)d_in[1];
  const int*   lens  = (const int*)d_in[2];
  const float* negk  = (const float*)d_in[3];
  const float* g_wih = (const float*)d_in[4];
  const float* g_whh = (const float*)d_in[5];
  const float* g_bih = (const float*)d_in[6];
  const float* g_bhh = (const float*)d_in[7];
  const float* a_wxr = (const float*)d_in[8];
  const float* a_whr = (const float*)d_in[9];
  const float* a_br  = (const float*)d_in[10];
  const float* a_wxh = (const float*)d_in[11];
  const float* a_whh = (const float*)d_in[12];
  const float* a_bh  = (const float*)d_in[13];
  const float* a_wxz = (const float*)d_in[14];
  const float* a_whz = (const float*)d_in[15];
  const float* a_bz  = (const float*)d_in[16];
  const float* t_W0  = (const float*)d_in[17];
  const float* t_b0  = (const float*)d_in[18];
  const float* t_a0  = (const float*)d_in[19];
  const float* t_W1  = (const float*)d_in[20];
  const float* t_b1  = (const float*)d_in[21];
  const float* t_a1  = (const float*)d_in[22];
  const float* t_W2  = (const float*)d_in[23];
  const float* t_b2  = (const float*)d_in[24];
  const float* x_W0  = (const float*)d_in[25];
  const float* x_b0  = (const float*)d_in[26];
  const float* x_W1  = (const float*)d_in[27];
  const float* x_b1  = (const float*)d_in[28];
  const float* x_W2  = (const float*)d_in[29];
  const float* x_b2  = (const float*)d_in[30];
  float* out = (float*)d_out;

  // Workspace carve (all offsets 256B aligned).
  char* ws = (char*)d_ws;
  size_t cur = 0;
  auto take = [&](size_t bytes) {
    size_t o = cur; cur += (bytes + 255) & ~(size_t)255; return o;
  };
  float*  interests = (float*)(ws + take(sizeof(float) * (size_t)Bb * Tt * Dd));
  float*  scores    = (float*)(ws + take(sizeof(float) * (size_t)Bb * Tt));
  __bf16* wih_p  = (__bf16*)(ws + take(2 * N3 * Dd));
  __bf16* whh_p  = (__bf16*)(ws + take(2 * N3 * Dd));
  __bf16* wx_p   = (__bf16*)(ws + take(2 * N3 * Dd));
  __bf16* whrz_p = (__bf16*)(ws + take(2 * 256 * Dd));
  __bf16* whhA_p = (__bf16*)(ws + take(2 * Dd * Dd));
  __bf16* w0x_p  = (__bf16*)(ws + take(2 * 64 * 256));
  float*  accum  = (float*)(ws + take(256));

  // --- weight packing ---
  pack_conv<<<(N3 * Dd + 255) / 256, 256, 0, stream>>>(wih_p, g_wih, N3 * Dd);
  pack_conv<<<(N3 * Dd + 255) / 256, 256, 0, stream>>>(whh_p, g_whh, N3 * Dd);
  int gT = (Dd * Dd + 255) / 256;
  pack_tr<<<gT, 256, 0, stream>>>(wx_p,               a_wxr, Dd, Dd);
  pack_tr<<<gT, 256, 0, stream>>>(wx_p + Dd * Dd,     a_wxz, Dd, Dd);
  pack_tr<<<gT, 256, 0, stream>>>(wx_p + 2 * Dd * Dd, a_wxh, Dd, Dd);
  pack_tr<<<gT, 256, 0, stream>>>(whrz_p,             a_whr, Dd, Dd);
  pack_tr<<<gT, 256, 0, stream>>>(whrz_p + Dd * Dd,   a_whz, Dd, Dd);
  pack_tr<<<gT, 256, 0, stream>>>(whhA_p,             a_whh, Dd, Dd);
  pack_tr<<<(64 * 256 + 255) / 256, 256, 0, stream>>>(w0x_p, x_W0, 256, 64);
  init_accum<<<1, 1, 0, stream>>>(accum);

  // --- pipeline ---
  constexpr size_t GRU_LDS =
      2 * (size_t)(2 * N3 * Dd) +                // wih + whh (bf16)
      2 * (size_t)(2 * 16 * Dd) +                // xbf + hbf (bf16)
      4 * (size_t)(2 * 16 * N3 + 16 * Dd);       // gi + gh + hs (f32)
  gru_fused<<<Bb / 16, 256, GRU_LDS, stream>>>(keys, lens, wih_p, whh_p,
                                               g_bih, g_bhh, interests);

  attention_k<<<Bb, 256, 0, stream>>>(query, interests, lens, t_W0, t_b0, t_a0,
                                      t_W1, t_b1, t_a1, t_W2, t_b2, scores);

  aux_k<<<(Bb * (Tt - 1)) / 16, 128, 0, stream>>>(interests, keys, negk, lens,
                                                  w0x_p, x_b0, x_W1, x_b1,
                                                  x_W2, x_b2, accum);

  constexpr size_t AUG_LDS =
      (size_t)2 * (N3 * Dd + 256 * Dd + Dd * Dd) +    // wx + whrz + whh (bf16)
      (size_t)2 * (3 * 16 * Dd) +                     // xbf + hbf + rhbf (bf16)
      (size_t)4 * (16 * N3 + 16 * 256 + 2 * 16 * Dd); // gx + gh + hs + zs (f32)
  augru_fused<<<Bb / 16, 256, AUG_LDS, stream>>>(interests, scores, lens, wx_p,
                                                 whrz_p, whhA_p, a_br, a_bz,
                                                 a_bh, out);

  finalize_k<<<1, 1, 0, stream>>>(accum, lens, out + (size_t)Bb * Dd);
}
